// SResnet_35167192220212
// MI455X (gfx1250) — matrix operand order repaired and checked by reading, not compiled, not run on hardware
//
#include <hip/hip_runtime.h>
#include <cstddef>

typedef __attribute__((ext_vector_type(16))) _Float16 v16h;
typedef __attribute__((ext_vector_type(8)))  float    v8f;
typedef unsigned int v4u __attribute__((ext_vector_type(4)));
typedef int          v8i __attribute__((ext_vector_type(8)));
typedef int          v4i __attribute__((ext_vector_type(4)));

#define LEAK 0.95f
#define T_STEPS 5

// ---------------------------------------------------------------------------
// TDM: DMA a 2D weight tile (64 rows x 32 halves, row stride Kp halves) from
// global memory into LDS at lds_off. Descriptor per CDNA5 ISA ch.8:
//  group0: [1:0]=count=1, [63:32]=lds_addr, [120:64]=global_addr, [127:126]=2
//  group1: data_size=1 (2B), tensor_dim0=Kp, tensor_dim1=Cout,
//          tile_dim0=32, tile_dim1=64, tensor_dim0_stride=Kp
//  groups 2/3: zero (2-D tile, no iterate/gather)
// All operands uniform -> SGPR descriptor groups. TDM ignores EXEC; caller
// gates to one wave and pairs with s_wait_tensorcnt + workgroup barrier.
// ---------------------------------------------------------------------------
__device__ __forceinline__ void tdm_load_b_tile(unsigned lds_off, const void* gaddr,
                                                int Kp, int Cout) {
  unsigned long long ga = (unsigned long long)gaddr;
  v4u g0;
  g0[0] = 1u;                                          // count=1 (valid user D#)
  g0[1] = lds_off;                                     // LDS byte address
  g0[2] = (unsigned)(ga & 0xffffffffull);              // global_addr[31:0]
  g0[3] = (unsigned)((ga >> 32) & 0x1ffffffull) | (2u << 30);  // [56:32] | type=2
  v8i g1;
  g1[0] = (int)(1u << 16);                             // data_size=1 -> 2 bytes
  g1[1] = (int)(((unsigned)Kp & 0xffffu) << 16);       // tensor_dim0[15:0]
  g1[2] = (int)(((unsigned)Kp >> 16) |                 // tensor_dim0[31:16]
                (((unsigned)Cout & 0xffffu) << 16));   // tensor_dim1[15:0]
  g1[3] = (int)(((unsigned)Cout >> 16) | (32u << 16)); // tensor_dim1[31:16], tile_dim0=32
  g1[4] = 64;                                          // tile_dim1=64, tile_dim2=0
  g1[5] = Kp;                                          // tensor_dim0_stride[31:0]
  g1[6] = 0;                                           // stride hi, dim1_stride lo
  g1[7] = 0;
  v4i z4 = {0, 0, 0, 0};
#if defined(__clang_major__) && (__clang_major__ >= 23)
  v8i z8 = {0, 0, 0, 0, 0, 0, 0, 0};
  __builtin_amdgcn_tensor_load_to_lds(g0, g1, z4, z4, z8, 0);
#else
  __builtin_amdgcn_tensor_load_to_lds(g0, g1, z4, z4, 0);
#endif
}

// ---------------------------------------------------------------------------
// Stage one 128x32 im2col A tile chunk into LDS (per-thread vector gather).
// 2 threads per row, 16 halves each; chunk kc lies within one (ky,kx) slab
// because 32 | Cin for every WMMA layer.
// ---------------------------------------------------------------------------
__device__ __forceinline__ void stage_a_tile(_Float16* __restrict__ ldsA,
                                             const _Float16* __restrict__ act,
                                             int kc, int nChunks,
                                             int H, int W, int Cin,
                                             int stride, int ks, int pad,
                                             int b, int yo, int xo,
                                             int arow, int aseg) {
  int k0  = kc << 5;
  int kyx = k0 / Cin;
  int ci  = (k0 % Cin) + (aseg << 4);
  int ky = kyx / ks, kx = kyx % ks;
  int iy = yo * stride + ky - pad;
  int ix = xo * stride + kx - pad;
  uint4 v0 = make_uint4(0u, 0u, 0u, 0u), v1 = v0;
  if (iy >= 0 && iy < H && ix >= 0 && ix < W) {
    const uint4* g = reinterpret_cast<const uint4*>(
        act + ((size_t)((b * H + iy) * W + ix)) * Cin + ci);
    v0 = g[0];
    v1 = g[1];
    if (kc + 1 < nChunks)                 // speculative next-chunk prefetch
      __builtin_prefetch((const char*)g + 64, 0, 1);
  }
  *reinterpret_cast<uint4*>(&ldsA[arow * 32 + (aseg << 4)])     = v0;
  *reinterpret_cast<uint4*>(&ldsA[arow * 32 + (aseg << 4) + 8]) = v1;
}

// ---------------------------------------------------------------------------
// Pack OIHW f32 weights -> [Cout][K] f16, K = (ky*KH+kx)*Cin + ci
// so that im2col rows are contiguous runs of ci for each (ky,kx).
// ---------------------------------------------------------------------------
__global__ void pack_weights_k(const float* __restrict__ w, _Float16* __restrict__ wp,
                               int Cout, int Cin, int KH) {
  int K = KH * KH * Cin;
  int idx = blockIdx.x * blockDim.x + threadIdx.x;
  if (idx >= Cout * K) return;
  int co = idx / K, k = idx % K;
  int kyx = k / Cin, ci = k % Cin;
  int ky = kyx / KH, kx = kyx % KH;
  wp[idx] = (_Float16)w[((co * Cin + ci) * KH + ky) * KH + kx];
}

// ---------------------------------------------------------------------------
// Layer 0: direct f32 conv, 3x3, pad 1, Cin=3 (tiny: 0.4% of total FLOPs).
// inp [64][3][32][32] NCHW, w [64][3][3][3], y0 [65536][64] (NHWC rows).
// ---------------------------------------------------------------------------
__global__ void conv0_direct_k(const float* __restrict__ inp, const float* __restrict__ w,
                               float* __restrict__ y0) {
  int idx = blockIdx.x * blockDim.x + threadIdx.x;     // m*64 + c
  int c = idx & 63, m = idx >> 6;
  int b = m >> 10, r = m & 1023, yy = r >> 5, xx = r & 31;
  float acc = 0.f;
  for (int ci = 0; ci < 3; ++ci)
    for (int ky = 0; ky < 3; ++ky) {
      int iy = yy + ky - 1;
      if (iy < 0 || iy > 31) continue;
      for (int kx = 0; kx < 3; ++kx) {
        int ix = xx + kx - 1;
        if (ix < 0 || ix > 31) continue;
        acc += w[((c * 3 + ci) * 3 + ky) * 3 + kx] *
               inp[((b * 3 + ci) * 32 + iy) * 32 + ix];
      }
    }
  y0[idx] = acc;
}

// ---------------------------------------------------------------------------
// Implicit-GEMM conv via WMMA f16 -> f32, double-buffered software pipeline.
//   out[M=B*Ho*Wo][Cout] = sum_K  A_im2col[M][K] * Wp[K][Cout]
// Workgroup tile: 128(M) x 64(N), 8 waves (4x2), each wave 32x32 = 2x2 WMMA
// frags of 16x16, K in chunks of 32.
//   A tile: per-thread im2col vector gather into LDS buffer nxt.
//   B tile: Tensor Data Mover DMA into LDS buffer nxt (wave 0 issues).
// While chunk kc computes from buffer cur, chunk kc+1 streams into nxt;
// one s_wait_tensorcnt + one barrier per iteration rotates the buffers.
// Requires: Cin % 32 == 0, M % 128 == 0, Cout % 64 == 0 (true for all layers).
// ---------------------------------------------------------------------------
__launch_bounds__(256)
__global__ void conv_wmma_k(const _Float16* __restrict__ act,
                            const _Float16* __restrict__ wp,
                            float* __restrict__ y,
                            int H, int W, int Cin, int Ho, int Wo, int Cout,
                            int stride, int ks, int pad, int Kp) {
  __shared__ __align__(64) _Float16 ldsA[2][128 * 32];   // [buf][m][k]
  __shared__ __align__(64) _Float16 ldsB[2][64 * 32];    // [buf][n][k] (B^T)

  const int tid  = threadIdx.x;
  const int lane = tid & 31;
  const int wave = tid >> 5;
  const int wy = wave & 3, wx = wave >> 2;            // 4 x 2 wave grid
  const int mBase = blockIdx.x * 128;
  const int nBase = blockIdx.y * 64;

  // A-tile loader mapping: 2 threads per row, 16 halves each
  const int arow = tid >> 1, aseg = tid & 1;
  const int m  = mBase + arow;
  const int hw = Ho * Wo;
  const int b  = m / hw;
  const int sp = m % hw;
  const int yo = sp / Wo, xo = sp % Wo;

  const _Float16* wp_blk = wp + (size_t)nBase * Kp;
  const unsigned ldsB_off[2] = {(unsigned)(size_t)(void*)&ldsB[0][0],
                                (unsigned)(size_t)(void*)&ldsB[1][0]};

  v8f acc[2][2] = {};
  const int nChunks = Kp >> 5;

  // ---- prologue: stage chunk 0 into buffer 0 ----
  if (wave == 0) tdm_load_b_tile(ldsB_off[0], wp_blk, Kp, Cout);
  stage_a_tile(&ldsA[0][0], act, 0, nChunks, H, W, Cin, stride, ks, pad,
               b, yo, xo, arow, aseg);
  if (wave == 0) __builtin_amdgcn_s_wait_tensorcnt(0);
  __syncthreads();

  for (int kc = 0; kc < nChunks; ++kc) {
    const int cur = kc & 1, nxt = cur ^ 1;
    const bool more = (kc + 1 < nChunks);

    // ---- stage chunk kc+1 into the other buffer while computing kc ----
    if (more) {
      if (wave == 0)
        tdm_load_b_tile(ldsB_off[nxt], wp_blk + ((kc + 1) << 5), Kp, Cout);
      stage_a_tile(&ldsA[nxt][0], act, kc + 1, nChunks, H, W, Cin, stride, ks,
                   pad, b, yo, xo, arow, aseg);
    }

    // ---- fragments (ISA 7.12.2 layouts) from buffer cur ----
    union Frag { v16h v; uint4 q[2]; };
    Frag afr[2], bfr[2];
    // A 16x32: lanes 0-15 row M=lane, K {0..7,16..23}; lanes 16-31 K {8..15,24..31}
    const int ak0 = (lane < 16) ? 0 : 8;
#pragma unroll
    for (int mi = 0; mi < 2; ++mi) {
      int rrow = wy * 32 + mi * 16 + (lane & 15);
      afr[mi].q[0] = *reinterpret_cast<const uint4*>(&ldsA[cur][rrow * 32 + ak0]);
      afr[mi].q[1] = *reinterpret_cast<const uint4*>(&ldsA[cur][rrow * 32 + ak0 + 16]);
    }
    // B 32x16: lanes 0-15 col N=lane, K {0..7,16..23}; lanes 16-31 K {8..15,24..31}
    const int bk0 = (lane < 16) ? 0 : 8;
#pragma unroll
    for (int ni = 0; ni < 2; ++ni) {
      int nrow = wx * 32 + ni * 16 + (lane & 15);
      bfr[ni].q[0] = *reinterpret_cast<const uint4*>(&ldsB[cur][nrow * 32 + bk0]);
      bfr[ni].q[1] = *reinterpret_cast<const uint4*>(&ldsB[cur][nrow * 32 + bk0 + 16]);
    }
#pragma unroll
    for (int mi = 0; mi < 2; ++mi)
#pragma unroll
      for (int ni = 0; ni < 2; ++ni)
        acc[mi][ni] = __builtin_amdgcn_wmma_f32_16x16x32_f16(
            false, afr[mi].v, false, bfr[ni].v, (short)0, acc[mi][ni],
            false, false);

    // next TDM must have landed before anyone reads nxt after the barrier
    if (more && wave == 0) __builtin_amdgcn_s_wait_tensorcnt(0);
    __syncthreads();
  }

  // ---- store C (16x16 f32 layout: VGPR r -> M=r (+8 for hi lanes), lane -> N)
#pragma unroll
  for (int mi = 0; mi < 2; ++mi)
#pragma unroll
    for (int ni = 0; ni < 2; ++ni) {
      int col   = nBase + wx * 32 + ni * 16 + (lane & 15);
      int rbase = mBase + wy * 32 + mi * 16 + ((lane >> 4) << 3);
      float* yp = y + (size_t)rbase * Cout + col;
#pragma unroll
      for (int r = 0; r < 8; ++r)
        yp[(size_t)r * Cout] = acc[mi][ni][r];
    }
}

// ---------------------------------------------------------------------------
// Training-mode BN statistics per channel (biased var over M = B*H*W).
// One block per channel. Writes (mean, rsqrt(var+eps)).
// ---------------------------------------------------------------------------
__global__ void bn_stats_k(const float* __restrict__ y, float2* __restrict__ stats,
                           int M, int C) {
  __shared__ float ssum[256], ssq[256];
  int c = blockIdx.x, t = threadIdx.x;
  float s = 0.f, q = 0.f;
  for (int m = t; m < M; m += 256) {
    float v = y[(size_t)m * C + c];
    s += v;
    q += v * v;
  }
  ssum[t] = s; ssq[t] = q;
  __syncthreads();
  for (int st = 128; st > 0; st >>= 1) {
    if (t < st) { ssum[t] += ssum[t + st]; ssq[t] += ssq[t + st]; }
    __syncthreads();
  }
  if (t == 0) {
    float mean = ssum[0] / (float)M;
    float var  = ssq[0] / (float)M - mean * mean;
    stats[c] = make_float2(mean, rsqrtf(var + 1e-4f));
  }
}

// ---------------------------------------------------------------------------
// Fused BN-apply + LIF membrane update + spike + reset + residual add.
// mode 0: out = spike
// mode 1: out = spike + skip (f16)
// mode 2: out = spike + BN_res(yres)  (1x1-conv residual path with beta)
// ---------------------------------------------------------------------------
__global__ void lif_k(const float* __restrict__ y, const float2* __restrict__ stats,
                      const float* __restrict__ gamma_t,
                      float* __restrict__ mem, _Float16* __restrict__ out,
                      const _Float16* __restrict__ skip,
                      const float* __restrict__ yres, const float2* __restrict__ stats_r,
                      const float* __restrict__ gr_t, const float* __restrict__ br_t,
                      int total, int C, int mode) {
  int idx = blockIdx.x * blockDim.x + threadIdx.x;
  if (idx >= total) return;
  int c = idx % C;
  float2 st  = stats[c];
  float yhat = (y[idx] - st.x) * st.y * gamma_t[c];
  float mnew = LEAK * mem[idx] + yhat;           // mem zero-init => t=0 gives yhat
  float thr  = mnew - 1.0f;                      // mem/THR - 1, THR = 1
  float spk  = (thr > 0.f) ? 1.f : 0.f;
  mem[idx]   = mnew - spk;                       // subtract THR where fired
  float o = spk;
  if (mode == 1) {
    o += (float)skip[idx];
  } else if (mode == 2) {
    float2 sr = stats_r[c];
    o += (yres[idx] - sr.x) * sr.y * gr_t[c] + br_t[c];
  }
  out[idx] = (_Float16)o;
}

// ---------------------------------------------------------------------------
// AdaptiveAvgPool(1,1) over act[64][8*8][256] + FC accumulate into mem_fc.
// ---------------------------------------------------------------------------
__global__ void pool_fc_k(const _Float16* __restrict__ act, const float* __restrict__ fcw,
                          float* __restrict__ mem_fc) {
  __shared__ float pooled[256];
  int b = blockIdx.x, t = threadIdx.x;
  float s = 0.f;
  for (int sp = 0; sp < 64; ++sp)
    s += (float)act[((size_t)b * 64 + sp) * 256 + t];
  pooled[t] = s * (1.f / 64.f);
  __syncthreads();
  if (t < 10) {
    float d = 0.f;
    for (int c = 0; c < 256; ++c) d += pooled[c] * fcw[t * 256 + c];
    mem_fc[b * 10 + t] += d;
  }
}

__global__ void finalize_k(const float* __restrict__ mem_fc, float* __restrict__ out) {
  int i = blockIdx.x * blockDim.x + threadIdx.x;
  if (i < 640) out[i] = mem_fc[i] * (1.0f / (float)T_STEPS);
}

// ---------------------------------------------------------------------------
// Host launch sequence
// ---------------------------------------------------------------------------
extern "C" void kernel_launch(void* const* d_in, const int* in_sizes, int n_in,
                              void* d_out, int out_size, void* d_ws, size_t ws_size,
                              hipStream_t stream) {
  (void)in_sizes; (void)n_in; (void)out_size; (void)ws_size;

  const float* inp = (const float*)d_in[0];
  const float* conv_w[13];
  const float* bn_g[13];
  for (int i = 0; i < 13; ++i) {
    conv_w[i] = (const float*)d_in[1 + i];
    bn_g[i]   = (const float*)d_in[14 + i];
  }
  const float* res_w[2] = {(const float*)d_in[27], (const float*)d_in[28]};
  const float* res_g[2] = {(const float*)d_in[29], (const float*)d_in[30]};
  const float* res_b[2] = {(const float*)d_in[31], (const float*)d_in[32]};
  const float* fc_w     = (const float*)d_in[33];
  float* outp = (float*)d_out;

  struct LP { int Cin, H, W, Cout, Ho, Wo, stride, ks, pad; };
  const LP lp[13] = {
    {  3,32,32, 64,32,32,1,3,1},
    { 64,32,32, 64,32,32,1,3,1},{ 64,32,32, 64,32,32,1,3,1},
    { 64,32,32, 64,32,32,1,3,1},{ 64,32,32, 64,32,32,1,3,1},
    { 64,32,32,128,16,16,2,3,1},
    {128,16,16,128,16,16,1,3,1},{128,16,16,128,16,16,1,3,1},{128,16,16,128,16,16,1,3,1},
    {128,16,16,256, 8, 8,2,3,1},
    {256, 8, 8,256, 8, 8,1,3,1},{256, 8, 8,256, 8, 8,1,3,1},{256, 8, 8,256, 8, 8,1,3,1},
  };
  const LP rp[2] = {
    { 64,32,32,128,16,16,2,1,0},
    {128,16,16,256, 8, 8,2,1,0},
  };

  int M[13];
  for (int i = 0; i < 13; ++i) M[i] = 64 * lp[i].Ho * lp[i].Wo;
  const int Mr[2] = {64 * 16 * 16, 64 * 8 * 8};

  // ---- bump allocator over d_ws (deterministic layout every call) ----
  char* wbase = (char*)d_ws;
  size_t woff = 0;
  auto alloc = [&](size_t bytes) -> char* {
    woff = (woff + 255) & ~(size_t)255;
    char* p = wbase + woff;
    woff += bytes;
    return p;
  };

  _Float16* act[13];
  for (int i = 0; i < 13; ++i)
    act[i] = (_Float16*)alloc((size_t)M[i] * lp[i].Cout * 2);

  // persistent membrane state (f32) + mem_fc, contiguous so one memset clears all
  float* memP[13];
  for (int i = 0; i < 13; ++i)
    memP[i] = (float*)alloc((size_t)M[i] * lp[i].Cout * 4);
  float* memfc = (float*)alloc(640 * 4);
  size_t mem_span = (size_t)((char*)memfc + 640 * 4 - (char*)memP[0]);
  hipMemsetAsync(memP[0], 0, mem_span, stream);

  float*  y0    = (float*)alloc((size_t)65536 * 64 * 4);
  float*  ymain = (float*)alloc((size_t)65536 * 64 * 4);
  float*  yres  = (float*)alloc((size_t)16384 * 128 * 4);
  float2* stats0  = (float2*)alloc(64 * sizeof(float2));
  float2* stats_m = (float2*)alloc(256 * sizeof(float2));
  float2* stats_r = (float2*)alloc(256 * sizeof(float2));
  _Float16* wp[13]; wp[0] = nullptr;
  for (int i = 1; i < 13; ++i)
    wp[i] = (_Float16*)alloc((size_t)lp[i].Cout * 9 * lp[i].Cin * 2);
  _Float16* wpr[2];
  for (int r = 0; r < 2; ++r)
    wpr[r] = (_Float16*)alloc((size_t)rp[r].Cout * rp[r].Cin * 2);

  // ---- one-time: pack weights ----
  for (int i = 1; i < 13; ++i) {
    int n = lp[i].Cout * 9 * lp[i].Cin;
    pack_weights_k<<<(n + 255) / 256, 256, 0, stream>>>(conv_w[i], wp[i],
                                                        lp[i].Cout, lp[i].Cin, 3);
  }
  for (int r = 0; r < 2; ++r) {
    int n = rp[r].Cout * rp[r].Cin;
    pack_weights_k<<<(n + 255) / 256, 256, 0, stream>>>(res_w[r], wpr[r],
                                                        rp[r].Cout, rp[r].Cin, 1);
  }

  // ---- layer 0 conv + BN stats are t-invariant: hoist out of time loop ----
  conv0_direct_k<<<(65536 * 64) / 256, 256, 0, stream>>>(inp, conv_w[0], y0);
  bn_stats_k<<<64, 256, 0, stream>>>(y0, stats0, 65536, 64);

  for (int t = 0; t < T_STEPS; ++t) {
    // layer 0: LIF on the cached conv output
    {
      int total = M[0] * 64;
      lif_k<<<(total + 255) / 256, 256, 0, stream>>>(
          y0, stats0, bn_g[0] + t * 64, memP[0], act[0],
          nullptr, nullptr, nullptr, nullptr, nullptr, total, 64, 0);
    }
    const _Float16* skip = act[0];
    int ridx = 0;

    for (int i = 1; i < 13; ++i) {
      const LP& L = lp[i];
      // main 3x3 conv (implicit GEMM, WMMA f16, TDM weight staging)
      conv_wmma_k<<<dim3(M[i] / 128, L.Cout / 64), 256, 0, stream>>>(
          act[i - 1], wp[i], ymain,
          L.H, L.W, L.Cin, L.Ho, L.Wo, L.Cout, L.stride, L.ks, L.pad,
          9 * L.Cin);
      bn_stats_k<<<L.Cout, 256, 0, stream>>>(ymain, stats_m, M[i], L.Cout);

      int mode = 0;
      const _Float16* skp = nullptr;
      const float *yres_p = nullptr, *grp = nullptr, *brp = nullptr;
      const float2* srp = nullptr;

      if ((i % 2) == 0) {
        if (i == 6 || i == 10) {
          // residual 1x1 stride-2 conv + BN(gamma,beta) on the skip tensor
          const LP& R = rp[ridx];
          conv_wmma_k<<<dim3(Mr[ridx] / 128, R.Cout / 64), 256, 0, stream>>>(
              skip, wpr[ridx], yres,
              R.H, R.W, R.Cin, R.Ho, R.Wo, R.Cout, R.stride, R.ks, R.pad,
              R.Cin);
          bn_stats_k<<<R.Cout, 256, 0, stream>>>(yres, stats_r, Mr[ridx], R.Cout);
          mode = 2;
          yres_p = yres; srp = stats_r;
          grp = res_g[ridx] + t * R.Cout;
          brp = res_b[ridx] + t * R.Cout;
          ++ridx;
        } else {
          mode = 1;
          skp = skip;
        }
      }

      int total = M[i] * L.Cout;
      lif_k<<<(total + 255) / 256, 256, 0, stream>>>(
          ymain, stats_m, bn_g[i] + t * L.Cout, memP[i], act[i],
          skp, yres_p, srp, grp, brp, total, L.Cout, mode);

      if ((i % 2) == 0) skip = act[i];
    }

    pool_fc_k<<<64, 256, 0, stream>>>(act[12], fc_w, memfc);
  }

  finalize_k<<<3, 256, 0, stream>>>(memfc, outp);
}
